// MultiHeadAttention_6004364280459
// MI455X (gfx1250) — matrix operand order repaired and checked by reading, not compiled, not run on hardware
//
#include <hip/hip_runtime.h>

// MI455X / gfx1250 Transformer-XL attention, bf16 WMMA pipeline.
// - All matmuls (projections, QK^T, Q·R^T band, P·V, MLP) use
//   v_wmma_f32_16x16x32_bf16.
// - Flash-attention streaming: the 4x8x1024x2048 score tensor (256MB) is
//   never materialized.
// - K/V tiles are DMA'd into LDS by the Tensor Data Mover
//   (tensor_load_to_lds, TENSORcnt double-buffering): one wave issues the
//   descriptors, all 4 waves of the block consume the shared LDS tiles.

typedef __attribute__((ext_vector_type(16))) __bf16 v16bf;
typedef __attribute__((ext_vector_type(8)))  __bf16 v8bf;
typedef __attribute__((ext_vector_type(8)))  float  v8f;
typedef unsigned int u32x4 __attribute__((ext_vector_type(4)));
typedef int i32x4 __attribute__((ext_vector_type(4)));
typedef int i32x8 __attribute__((ext_vector_type(8)));

#if defined(__HIP_DEVICE_COMPILE__) && __has_builtin(__builtin_amdgcn_tensor_load_to_lds)
#define HAS_TDM 1
#else
#define HAS_TDM 0
#endif

__device__ __forceinline__ v8f wmma_bf16(v16bf a, v16bf b, v8f c) {
  // 8 args: (neg_a, A, neg_b, B, c_mod, C, reuse_a, reuse_b)
  return __builtin_amdgcn_wmma_f32_16x16x32_bf16(false, a, false, b, (short)0, c,
                                                 false, false);
}

// ---- Tensor Data Mover: 2-D tile of 2-byte elements, global -> LDS --------
// D# built per CDNA5 ISA ch.8: group0 = {count/type/addr}, group1 = dims.
__device__ __forceinline__ void tdm_load_2d(unsigned lds_bytes, const void* g,
                                            unsigned td0, unsigned td1,
                                            unsigned tl0, unsigned tl1,
                                            unsigned long long stride0) {
#if HAS_TDM
  unsigned long long ga = (unsigned long long)(uintptr_t)g;
  u32x4 g0;
  g0[0] = 1u;                                 // count=1, user descriptor
  g0[1] = lds_bytes;                          // lds_addr (bytes)
  g0[2] = (unsigned)ga;                       // global_addr[31:0]
  g0[3] = ((unsigned)(ga >> 32) & 0x01FFFFFFu) | 0x80000000u;  // addr hi, type=2
  i32x8 g1;
  g1[0] = (int)(1u << 16);                    // wg_mask=0, data_size=1 (2B)
  g1[1] = (int)((td0 & 0xFFFFu) << 16);       // tensor_dim0 lo16 @ bits 63:48
  g1[2] = (int)((td0 >> 16) | ((td1 & 0xFFFFu) << 16));  // td0 hi | td1 lo
  g1[3] = (int)((td1 >> 16) | (tl0 << 16));   // td1 hi | tile_dim0
  g1[4] = (int)tl1;                           // tile_dim1 | tile_dim2=0
  g1[5] = (int)(unsigned)stride0;             // tensor_dim0_stride lo32
  g1[6] = (int)(unsigned)(stride0 >> 32);     // stride0 hi16 | stride1 lo16=0
  g1[7] = 0;
  i32x4 gz = {0, 0, 0, 0};
#if defined(__clang_major__) && (__clang_major__ >= 23)
  i32x8 gz8 = {0, 0, 0, 0, 0, 0, 0, 0};
  __builtin_amdgcn_tensor_load_to_lds(g0, g1, gz, gz, gz8, 0);
#else
  __builtin_amdgcn_tensor_load_to_lds(g0, g1, gz, gz, 0);
#endif
#else
  (void)lds_bytes; (void)g; (void)td0; (void)td1; (void)tl0; (void)tl1;
  (void)stride0;
#endif
}

__device__ __forceinline__ void wait_tensorcnt0() {
#if defined(__HIP_DEVICE_COMPILE__) && __has_builtin(__builtin_amdgcn_s_wait_tensorcnt)
  __builtin_amdgcn_s_wait_tensorcnt(0);
#endif
}

__device__ __forceinline__ float red_max16(float x) {
  x = fmaxf(x, __shfl_xor(x, 1, 32));
  x = fmaxf(x, __shfl_xor(x, 2, 32));
  x = fmaxf(x, __shfl_xor(x, 4, 32));
  x = fmaxf(x, __shfl_xor(x, 8, 32));
  return x;
}
__device__ __forceinline__ float red_sum16(float x) {
  x += __shfl_xor(x, 1, 32);
  x += __shfl_xor(x, 2, 32);
  x += __shfl_xor(x, 4, 32);
  x += __shfl_xor(x, 8, 32);
  return x;
}
__device__ __forceinline__ float red_sum32(float x) {
  x += __shfl_xor(x, 1, 32);
  x += __shfl_xor(x, 2, 32);
  x += __shfl_xor(x, 4, 32);
  x += __shfl_xor(x, 8, 32);
  x += __shfl_xor(x, 16, 32);
  return x;
}

// ---------------------------------------------------------------- weights ---
// out[n*K + k] = (bf16) in[k*N + n]
__global__ __launch_bounds__(256) void wtrans(const float* __restrict__ in,
                                              __bf16* __restrict__ out, int K,
                                              int N) {
  int id = blockIdx.x * 256 + threadIdx.x;
  if (id >= K * N) return;
  int n = id / K;
  int k = id - n * K;
  out[id] = (__bf16)in[k * N + n];
}

// ------------------------------------------------------- sinusoid position ---
__device__ __forceinline__ float pos_val(int prow, int j) {
  // pos = sinusoid(TOTAL, 128)[::-1] -> row prow reads logical P = 2047-prow
  float P = (float)(2047 - prow);
  float ang = P * exp2f(-(float)j * 0.20762050595f);  // 10000^(-2j/128)
  return ((j & 1) == 0) ? __sinf(ang) : __cosf(ang);
}

// ----------------------------------------------------- projection GEMM K=128 -
// C(MxN, bf16) = A(Mx128, f32) @ W(128xN), WT = (NxK) bf16.
// mode 0: plain rows. mode 1: kv rows (mem for t<1024, x after). mode 2: pos.
__global__ __launch_bounds__(256) void proj_gemm_k128(
    const float* __restrict__ A, const float* __restrict__ A2,
    const __bf16* __restrict__ WT, __bf16* __restrict__ C, int M, int N,
    int mode) {
  int lane = threadIdx.x & 31;
  int wv = threadIdx.x >> 5;
  int wid = blockIdx.x * 8 + wv;
  int nt4 = N >> 6;
  int mt = wid / nt4;
  int nb = wid - mt * nt4;
  int m0 = mt << 4, n0 = nb << 6;
  int half = lane & 16;
  int nl = lane & 15;
  int row = m0 + nl;

  const float* ap = A;
  if (mode == 1) {
    int b = row >> 11, t = row & 2047;
    ap = (t < 1024) ? (A + (size_t)(b * 1024 + t) * 128)
                    : (A2 + (size_t)(b * 1024 + (t - 1024)) * 128);
  } else if (mode == 0) {
    ap = A + (size_t)row * 128;
  }

  v16bf afr[4];
#pragma unroll
  for (int kc = 0; kc < 4; ++kc) {
    int klo = kc * 32 + (half ? 16 : 0);
    v16bf f;
#pragma unroll
    for (int i = 0; i < 8; ++i) {
      float lo, hi;
      if (mode == 2) {
        lo = pos_val(row, klo + i);
        hi = pos_val(row, klo + 8 + i);
      } else {
        lo = ap[klo + i];
        hi = ap[klo + 8 + i];
      }
      f[i] = (__bf16)lo;
      f[8 + i] = (__bf16)hi;
    }
    afr[kc] = f;
  }

  v8f zero = {};
  v8f acc[4] = {zero, zero, zero, zero};
#pragma unroll
  for (int kc = 0; kc < 4; ++kc) {
    int kb = kc * 32 + (half ? 16 : 0);
#pragma unroll
    for (int j = 0; j < 4; ++j) {
      int ncol = n0 + j * 16 + nl;
      v16bf bf = *(const v16bf*)(WT + (size_t)ncol * 128 + kb);
      acc[j] = wmma_bf16(afr[kc], bf, acc[j]);
    }
  }
#pragma unroll
  for (int j = 0; j < 4; ++j) {
    int ncol = n0 + j * 16 + nl;
#pragma unroll
    for (int v = 0; v < 8; ++v) {
      int mr = m0 + v + (half ? 8 : 0);
      C[(size_t)mr * N + ncol] = (__bf16)acc[j][v];
    }
  }
}

// --------------------------------------------------------------- vT build ---
__global__ __launch_bounds__(256) void vtranspose(const __bf16* __restrict__ kv,
                                                  __bf16* __restrict__ vT) {
  size_t id = (size_t)blockIdx.x * 256 + threadIdx.x;  // 32*2048*128
  int e = (int)(id & 127);
  size_t rt = id >> 7;  // bh*2048 + t
  size_t bh = rt >> 11;
  size_t t = rt & 2047;
  vT[(bh * 128 + e) * 2048 + t] = kv[(65536 + rt) * 128 + e];
}

// ------------------------------------------------------------ u1·k, u2·r ----
__global__ __launch_bounds__(256) void u_reduce(
    const __bf16* __restrict__ kvbf, const __bf16* __restrict__ rbf,
    const float* __restrict__ u1, const float* __restrict__ u2,
    float* __restrict__ u1k, float* __restrict__ u2r) {
  int rowid = blockIdx.x * 8 + (threadIdx.x >> 5);  // 0..81919
  int lane = threadIdx.x & 31;
  const __bf16* src;
  const float* uv;
  float* dst;
  if (rowid < 65536) {
    int hh = (rowid >> 11) & 7;
    src = kvbf + (size_t)rowid * 128;
    uv = u1 + hh * 128;
    dst = u1k + rowid;
  } else {
    int rr = rowid - 65536;
    int hh = rr >> 11;
    src = rbf + (size_t)rr * 128;
    uv = u2 + hh * 128;
    dst = u2r + rr;
  }
  float s = 0.f;
#pragma unroll
  for (int i = 0; i < 4; ++i) {
    int e = lane * 4 + i;
    s += (float)src[e] * uv[e];
  }
  s = red_sum32(s);
  if (lane == 0) *dst = s;
}

// ------------------------------------------------------------ flash attn ----
// 512 blocks = (b*8+h)*16 + sblk ; 4 waves/block, each owns 16 query rows.
// K/V tiles double-buffered in LDS via TDM; R band loaded direct (per-wave
// diagonal). 28 WMMAs per 32-column tile per wave.
__global__ __launch_bounds__(128) void flash_attn(
    const __bf16* __restrict__ qbf, const __bf16* __restrict__ kbf,
    const __bf16* __restrict__ rbf, const __bf16* __restrict__ vT,
    const float* __restrict__ u1k, const float* __restrict__ u2r,
    __bf16* __restrict__ attbf) {
  __shared__ __align__(16) __bf16 kbuf[2][32][128];  // K tile: t x e
  __shared__ __align__(16) __bf16 vbuf[2][128][32];  // V tile: e x t
  __shared__ __align__(16) __bf16 pbuf[4][16][32];
  const float NEGINF = -__builtin_inff();
  int lane = threadIdx.x & 31;
  int wv = threadIdx.x >> 5;
  int half = lane & 16;
  int nl = lane & 15;
  int blk = blockIdx.x;
  int sblk = blk & 15;
  int bh = blk >> 4;  // b*8+h
  int h = bh & 7;
  int s0 = sblk << 6;
  int sw = s0 + (wv << 4);

  const __bf16* qrow = qbf + (size_t)(bh * 1024 + sw + nl) * 128;
  v16bf aq[4];
#pragma unroll
  for (int kc = 0; kc < 4; ++kc) {
    int klo = kc * 32 + (half ? 16 : 0);
    union { v16bf v; v8bf h2[2]; } u;
    u.h2[0] = *(const v8bf*)(qrow + klo);
    u.h2[1] = *(const v8bf*)(qrow + klo + 8);
    aq[kc] = u.v;
  }

  v8f zero = {};
  v8f acc[8];
#pragma unroll
  for (int i = 0; i < 8; ++i) acc[i] = zero;
  float rm[8], rl[8];
#pragma unroll
  for (int v = 0; v < 8; ++v) { rm[v] = NEGINF; rl[v] = 0.0f; }

  const __bf16* kbase = kbf + (size_t)bh * 2048 * 128;
  const __bf16* rbase = rbf + (size_t)h * 2048 * 128;
  const __bf16* vbase = vT + (size_t)bh * 128 * 2048;
  const float* u1b = u1k + bh * 2048;
  const float* u2b = u2r + h * 2048;
  const float scale = 0.08838834764831845f;  // 1/sqrt(128)

  int nblk = (s0 + 1088) >> 5;  // block-uniform: max valid t is s0+63+1024

#if HAS_TDM
  // prologue: DMA tile 0 into buffer 0 (one wave issues; TENSORcnt tracks)
  if (wv == 0) {
    tdm_load_2d((unsigned)(uintptr_t)&kbuf[0][0][0], kbase, 128u, 32u, 128u,
                32u, 128ull);
    tdm_load_2d((unsigned)(uintptr_t)&vbuf[0][0][0], vbase, 32u, 128u, 32u,
                128u, 2048ull);
  }
#endif

  for (int tb = 0; tb < nblk; ++tb) {
    int t0 = tb << 5;
#if HAS_TDM
    int cur = tb & 1;
    if (wv == 0) wait_tensorcnt0();  // tile tb resident in kbuf/vbuf[cur]
    __syncthreads();                 // publish to all waves
    if (wv == 0 && tb + 1 < nblk) {  // DMA next tile while computing this one
      int t1 = (tb + 1) << 5;
      tdm_load_2d((unsigned)(uintptr_t)&kbuf[cur ^ 1][0][0],
                  kbase + (size_t)t1 * 128, 128u, 32u, 128u, 32u, 128ull);
      tdm_load_2d((unsigned)(uintptr_t)&vbuf[cur ^ 1][0][0], vbase + t1, 32u,
                  128u, 32u, 128u, 2048ull);
    }
#else
    int cur = 0;
    __syncthreads();
    // fallback: cooperative global->LDS copy
    for (int i = threadIdx.x; i < 512; i += 128)
      *(v8bf*)(&kbuf[0][0][0] + i * 8) =
          *(const v8bf*)(kbase + (size_t)t0 * 128 + i * 8);
    for (int i = threadIdx.x; i < 512; i += 128) {
      int er = i >> 2, tc = (i & 3) * 8;
      *(v8bf*)(&vbuf[0][er][tc]) =
          *(const v8bf*)(vbase + (size_t)er * 2048 + t0 + tc);
    }
    __syncthreads();
#endif

    // ---- QK^T: S tile 16x32 (K operands from LDS) ----
    const __bf16* kt = &kbuf[cur][0][0];
    const __bf16* vt = &vbuf[cur][0][0];
    v8f S0 = zero, S1 = zero;
#pragma unroll
    for (int kc = 0; kc < 4; ++kc) {
      int kb = kc * 32 + (half ? 16 : 0);
      v16bf b0 = *(const v16bf*)(kt + (size_t)nl * 128 + kb);
      v16bf b1 = *(const v16bf*)(kt + (size_t)(nl + 16) * 128 + kb);
      S0 = wmma_bf16(aq[kc], b0, S0);
      S1 = wmma_bf16(aq[kc], b1, S1);
    }
    // ---- relative-position band: W[m][c] = q[sw+m]·r[jb+c], c in [0,48) ----
    // b[s,t] = q[s]·r[t-s+1023]; no mod-wrap inside the unmasked region.
    int jb = t0 - sw + 1008;  // = t0 - sw + 1023 - 15
    int j0 = jb + nl;       j0 = j0 < 0 ? 0 : (j0 > 2047 ? 2047 : j0);
    int j1 = jb + 16 + nl;  j1 = j1 < 0 ? 0 : (j1 > 2047 ? 2047 : j1);
    int j2 = jb + 32 + nl;  j2 = j2 < 0 ? 0 : (j2 > 2047 ? 2047 : j2);
    int jp = j2 + 32; jp = jp > 2047 ? 2047 : jp;
    __builtin_prefetch(rbase + (size_t)jp * 128, 0, 1);
    v8f W0 = zero, W1 = zero, W2 = zero;
#pragma unroll
    for (int kc = 0; kc < 4; ++kc) {
      int kb = kc * 32 + (half ? 16 : 0);
      W0 = wmma_bf16(aq[kc], *(const v16bf*)(rbase + (size_t)j0 * 128 + kb), W0);
      W1 = wmma_bf16(aq[kc], *(const v16bf*)(rbase + (size_t)j1 * 128 + kb), W1);
      W2 = wmma_bf16(aq[kc], *(const v16bf*)(rbase + (size_t)j2 * 128 + kb), W2);
    }
    float ua = u1b[t0 + nl] + u2b[t0 + nl];
    float ub = u1b[t0 + 16 + nl] + u2b[t0 + 16 + nl];

    // ---- score assembly, mask, online softmax ----
    float p0[8], p1[8], csc[8];
#pragma unroll
    for (int v = 0; v < 8; ++v) {
      int m = v + (half >> 1);  // +8 on upper half-wave
      int c = nl + 15 - m;      // diagonal gather column, 0..30
      int sl = half | (c & 15);
      float a0 = __shfl(W0[v], sl, 32);
      float a1 = __shfl(W1[v], sl, 32);
      float a2 = __shfl(W2[v], sl, 32);
      float bt0 = (c < 16) ? a0 : a1;
      float bt1 = (c < 16) ? a1 : a2;
      int srow = sw + m;
      float r0 = S0[v] + bt0 + ua;
      float r1 = S1[v] + bt1 + ub;
      // reference: where(t-s<=1024, score, 0)/sqrt(d); where(==0, -inf)
      float sc0 = ((t0 + nl) - srow <= 1024) ? r0 * scale : 0.0f;
      float sc1 = ((t0 + 16 + nl) - srow <= 1024) ? r1 * scale : 0.0f;
      sc0 = (sc0 == 0.0f) ? NEGINF : sc0;
      sc1 = (sc1 == 0.0f) ? NEGINF : sc1;
      float tm = red_max16(fmaxf(sc0, sc1));
      float nm = fmaxf(rm[v], tm);
      float e0 = __expf(sc0 - nm);
      float e1 = __expf(sc1 - nm);
      float ts = red_sum16(e0 + e1);
      float corr = __expf(rm[v] - nm);
      rl[v] = rl[v] * corr + ts;
      rm[v] = nm;
      csc[v] = corr;
      p0[v] = e0;
      p1[v] = e1;
    }
    // ---- stage P through LDS: C-layout -> A-layout (bf16) ----
#pragma unroll
    for (int v = 0; v < 8; ++v) {
      int m = v + (half >> 1);
      pbuf[wv][m][nl] = (__bf16)p0[v];
      pbuf[wv][m][nl + 16] = (__bf16)p1[v];
    }
    __syncthreads();
    union { v16bf v; v8bf h2[2]; } pa;
    {
      int klo = half ? 16 : 0;
      pa.h2[0] = *(const v8bf*)(&pbuf[wv][nl][klo]);
      pa.h2[1] = *(const v8bf*)(&pbuf[wv][nl][klo + 8]);
    }
    // ---- rescale accumulator and P·V (V operand from LDS) ----
#pragma unroll
    for (int ec = 0; ec < 8; ++ec) {
#pragma unroll
      for (int v = 0; v < 8; ++v) acc[ec][v] *= csc[v];
      int tk = half ? 16 : 0;
      v16bf bv = *(const v16bf*)(vt + (size_t)(ec * 16 + nl) * 32 + tk);
      acc[ec] = wmma_bf16(pa.v, bv, acc[ec]);
    }
  }

  // epilogue: normalize and store att (flat (b,h,s,e) == reference reshape)
#pragma unroll
  for (int ec = 0; ec < 8; ++ec) {
#pragma unroll
    for (int v = 0; v < 8; ++v) {
      int m = v + (half >> 1);
      size_t o = (size_t)(bh * 1024 + sw + m) * 128 + ec * 16 + nl;
      attbf[o] = (__bf16)(acc[ec][v] / rl[v]);
    }
  }
}

// ------------------------------------------------------- MLP GEMM + resid ---
// Y(4096x128,f32) = att(4096x1024,bf16) @ w_mlp + x
__global__ __launch_bounds__(256) void mlp_gemm(const __bf16* __restrict__ Abf,
                                                const __bf16* __restrict__ WT,
                                                const float* __restrict__ X,
                                                float* __restrict__ Y) {
  int lane = threadIdx.x & 31;
  int wv = threadIdx.x >> 5;
  int wid = blockIdx.x * 8 + wv;  // 512 waves
  int mt = wid >> 1, nh = wid & 1;
  int m0 = mt << 4, n0 = nh << 6;
  int half = lane & 16;
  int nl = lane & 15;
  int row = m0 + nl;
  v8f zero = {};
  v8f acc[4] = {zero, zero, zero, zero};
  for (int kc = 0; kc < 32; ++kc) {
    int klo = kc * 32 + (half ? 16 : 0);
    union { v16bf v; v8bf h2[2]; } a;
    a.h2[0] = *(const v8bf*)(Abf + (size_t)row * 1024 + klo);
    a.h2[1] = *(const v8bf*)(Abf + (size_t)row * 1024 + klo + 8);
    int kb = kc * 32 + (half ? 16 : 0);
#pragma unroll
    for (int j = 0; j < 4; ++j) {
      int ncol = n0 + j * 16 + nl;
      v16bf bf = *(const v16bf*)(WT + (size_t)ncol * 1024 + kb);
      acc[j] = wmma_bf16(a.v, bf, acc[j]);
    }
  }
#pragma unroll
  for (int j = 0; j < 4; ++j) {
    int ncol = n0 + j * 16 + nl;
#pragma unroll
    for (int v = 0; v < 8; ++v) {
      int mr = m0 + v + (half ? 8 : 0);
      Y[(size_t)mr * 128 + ncol] = acc[j][v] + X[(size_t)mr * 128 + ncol];
    }
  }
}

// --------------------------------------------------------------- layernorm --
__global__ __launch_bounds__(256) void layernorm_k(const float* __restrict__ Y,
                                                   const float* __restrict__ g,
                                                   const float* __restrict__ bb,
                                                   float* __restrict__ out) {
  int row = blockIdx.x * 8 + (threadIdx.x >> 5);
  int lane = threadIdx.x & 31;
  float4 v = *(const float4*)(Y + (size_t)row * 128 + lane * 4);
  float s = red_sum32(v.x + v.y + v.z + v.w);
  float mu = s * 0.0078125f;
  float dx = v.x - mu, dy = v.y - mu, dz = v.z - mu, dw = v.w - mu;
  float q = red_sum32(dx * dx + dy * dy + dz * dz + dw * dw);
  float inv = rsqrtf(q * 0.0078125f + 1e-5f);
  float4 gg = *(const float4*)(g + lane * 4);
  float4 bv = *(const float4*)(bb + lane * 4);
  float4 o;
  o.x = dx * inv * gg.x + bv.x;
  o.y = dy * inv * gg.y + bv.y;
  o.z = dz * inv * gg.z + bv.z;
  o.w = dw * inv * gg.w + bv.w;
  *(float4*)(out + (size_t)row * 128 + lane * 4) = o;
}

// ------------------------------------------------------------------ launch --
extern "C" void kernel_launch(void* const* d_in, const int* in_sizes, int n_in,
                              void* d_out, int out_size, void* d_ws,
                              size_t ws_size, hipStream_t stream) {
  const float* x = (const float*)d_in[0];
  const float* mem = (const float*)d_in[1];
  // d_in[2] = att_mask (all zeros, unused by the reference math)
  const float* u1 = (const float*)d_in[3];
  const float* u2 = (const float*)d_in[4];
  const float* w_q = (const float*)d_in[5];
  const float* w_kv = (const float*)d_in[6];
  const float* w_r = (const float*)d_in[7];
  const float* w_mlp = (const float*)d_in[8];
  const float* ln_g = (const float*)d_in[9];
  const float* ln_b = (const float*)d_in[10];

  char* ws = (char*)d_ws;  // total ~75 MB
  __bf16* qbf = (__bf16*)(ws);                    // 4096x1024  bf16  8388608
  __bf16* kvbf = (__bf16*)(ws + 8388608);         // 8192x2048  bf16 33554432
  __bf16* rbf = (__bf16*)(ws + 41943040);         // 2048x1024  bf16  4194304
  __bf16* vTb = (__bf16*)(ws + 46137344);         // 32x128x2048 bf16 16777216
  __bf16* wqT = (__bf16*)(ws + 62914560);         // 1024x128         262144
  __bf16* wkvT = (__bf16*)(ws + 63176704);        // 2048x128         524288
  __bf16* wrT = (__bf16*)(ws + 63700992);         // 1024x128         262144
  __bf16* wmlpT = (__bf16*)(ws + 63963136);       // 128x1024         262144
  float* u1kB = (float*)(ws + 64225280);          // 4x8x2048 f32     262144
  float* u2rB = (float*)(ws + 64487424);          // 8x2048 f32        65536
  __bf16* attbf = (__bf16*)(ws + 64552960);       // 4096x1024 bf16  8388608
  float* ybuf = (float*)(ws + 72941568);          // 4096x128 f32    2097152

  // 1) weight convert+transpose (bf16, N-major)
  wtrans<<<512, 256, 0, stream>>>(w_q, wqT, 128, 1024);
  wtrans<<<1024, 256, 0, stream>>>(w_kv, wkvT, 128, 2048);
  wtrans<<<512, 256, 0, stream>>>(w_r, wrT, 128, 1024);
  wtrans<<<512, 256, 0, stream>>>(w_mlp, wmlpT, 1024, 128);

  // 2) projections (WMMA bf16, K=128)
  proj_gemm_k128<<<512, 256, 0, stream>>>(x, nullptr, wqT, qbf, 4096, 1024, 0);
  proj_gemm_k128<<<2048, 256, 0, stream>>>(mem, x, wkvT, kvbf, 8192, 2048, 1);
  proj_gemm_k128<<<256, 256, 0, stream>>>(nullptr, nullptr, wrT, rbf, 2048,
                                          1024, 2);

  // 3) v transpose for P·V B-operands; u1·k / u2·r bias rows
  vtranspose<<<32768, 256, 0, stream>>>(kvbf, vTb);
  u_reduce<<<10240, 256, 0, stream>>>(kvbf, rbf, u1, u2, u1kB, u2rB);

  // 4) flash attention (TDM double-buffered K/V, 28 WMMAs per 32-col tile)
  flash_attn<<<512, 128, 0, stream>>>(qbf, kvbf, rbf, vTb, u1kB, u2rB, attbf);

  // 5) output projection + residual, then layernorm
  mlp_gemm<<<64, 256, 0, stream>>>(attbf, wmlpT, x, ybuf);
  layernorm_k<<<512, 256, 0, stream>>>(ybuf, ln_g, ln_b, (float*)d_out);
}
